// SemanticPartDiscovery_45672682225741
// MI455X (gfx1250) — matrix operand order repaired and checked by reading, not compiled, not run on hardware
//
#include <hip/hip_runtime.h>
#include <math.h>

typedef __attribute__((ext_vector_type(16))) _Float16 v16h;
typedef __attribute__((ext_vector_type(8)))  float    v8f;

#define B_   64
#define N_   4096
#define DF   384
#define DP   256
#define KP   8
#define ROWS 32
#define TPB  256
#define KSTEPS (DF / 32)

// d_out offsets (floats): part_feat, part_pos, assign, salience, feat
#define PF_OFF   0
#define POS_OFF  131072
#define AS_OFF   132096
#define SAL_OFF  2229248
#define FEAT_OFF 2229760

// d_ws offsets (floats)
#define WS_MASS  0
#define WS_POS   512
#define WS_FNUM  1536
#define WS_ACC_TOTAL 132608          // zeroed region (floats)
#define WS_WH    132608              // f16 packed W_proj starts here (float offset)
#define WH_HALFS (DF * DP)           // 98304 halfs

__device__ __forceinline__ float gelu_exact(float x) {
    return 0.5f * x * (1.0f + erff(x * 0.70710678118654752f));
}

__device__ __forceinline__ float wave_allreduce(float v) {
    #pragma unroll
    for (int off = 16; off > 0; off >>= 1) v += __shfl_xor(v, off, 32);
    return v;
}

struct alignas(16) H8 { _Float16 h[8]; };

__global__ __launch_bounds__(256)
void zero_ws_kernel(float* __restrict__ ws) {
    int i = blockIdx.x * blockDim.x + threadIdx.x;
    if (i < WS_ACC_TOTAL) ws[i] = 0.0f;
}

// -----------------------------------------------------------------------------
// One-time pack: W_proj (f32 [384][256]) -> f16 in WMMA B-fragment layout:
//   Wh[step][coltile][lane][e], step=k/32, coltile=n/16,
//   lane=(n%16)+16*((k%32)>=16), e=k%16.
// 192 KB result stays resident in the 192 MB L2 for all 8192 consumer WGs.
// -----------------------------------------------------------------------------
__global__ __launch_bounds__(256)
void pack_w_kernel(const float* __restrict__ Wproj, float* __restrict__ ws) {
    _Float16* Wh = reinterpret_cast<_Float16*>(ws + WS_WH);
    int idx = blockIdx.x * blockDim.x + threadIdx.x;   // over 384*256
    if (idx < DF * DP) {
        int k = idx / DP, n = idx % DP;
        int step = k >> 5, kl = k & 31;
        size_t dst = (((size_t)step * 16 + (n >> 4)) * 32 + ((n & 15) + (kl & 16))) * 16
                     + (kl & 15);
        Wh[dst] = (_Float16)Wproj[(size_t)k * DP + n];
    }
}

// -----------------------------------------------------------------------------
// Fused: GEMM1 (WMMA f16->f32) + bias + LayerNorm + GELU + L2norm + proto sim +
// softmax + assign-weighted accumulation (mass / pos / feat numerators).
// One workgroup = 32 rows x 256 cols, 8 waves.
// -----------------------------------------------------------------------------
__global__ __launch_bounds__(TPB)
void fused_feat_kernel(const float* __restrict__ Ag_all,
                       const int* __restrict__ Hp, const int* __restrict__ Wp,
                       const _Float16* __restrict__ Wh, const float* __restrict__ bproj,
                       const float* __restrict__ g1, const float* __restrict__ beta1,
                       const float* __restrict__ proto,
                       float* __restrict__ out, float* __restrict__ ws)
{
    __shared__ _Float16 Atile[ROWS][32];        //  2 KB  (A chunk, f16)
    __shared__ _Float16 Bt[16][32][16];         // 16 KB  (W chunk, fragment layout)
    __shared__ float    featLDS[ROWS][DP];      // 32 KB
    __shared__ float    protoLDS[KP][DP];       //  8 KB
    __shared__ float    normLDS[KP];
    __shared__ float    biasLDS[DP];
    __shared__ float    g1LDS[DP];
    __shared__ float    b1LDS[DP];
    __shared__ float    assignLDS[ROWS][KP];    //  1 KB

    const int t    = threadIdx.x;
    const int lane = t & 31;
    const int wv   = t >> 5;
    const int blk  = blockIdx.x;
    const int b    = blk / (N_ / ROWS);
    const int n0   = (blk % (N_ / ROWS)) * ROWS;
    const int Hh   = Hp[0];
    const int Ww   = Wp[0];
    const float inv_w = (Ww > 1) ? 1.0f / (float)(Ww - 1) : 1.0f;
    const float inv_h = (Hh > 1) ? 1.0f / (float)(Hh - 1) : 1.0f;

    // ---- constants into LDS; normalize prototypes ----
    biasLDS[t] = bproj[t];
    g1LDS[t]   = g1[t];
    b1LDS[t]   = beta1[t];
    #pragma unroll
    for (int i = 0; i < KP; ++i) protoLDS[i][t] = proto[i * DP + t];
    __syncthreads();
    if (t < KP) {
        float s = 0.f;
        for (int d = 0; d < DP; ++d) { float v = protoLDS[t][d]; s += v * v; }
        normLDS[t] = fmaxf(sqrtf(s), 1e-12f);
    }
    __syncthreads();
    #pragma unroll
    for (int i = 0; i < KP; ++i) protoLDS[i][t] /= normLDS[i];

    // ---- GEMM: feat_pre = A(32x384) @ Wproj(384x256) ----
    const float* Ag = Ag_all + ((size_t)(b * N_ + n0)) * DF;

    const int rblk   = wv >> 2;          // 0..1   row tile
    const int ctbase = (wv & 3) * 4;     // 0,4,8,12 first col tile
    const int m      = lane & 15;
    const int khalf  = (lane >> 4) << 3; // 0 or 8 (A-layout half select)

    v8f acc[4] = {};

    for (int k0 = 0; k0 < DF; k0 += 32) {
        __syncthreads();
        // stage A chunk (32 rows x 32 k), f32 -> f16
        #pragma unroll
        for (int i = 0; i < 4; ++i) {
            int e = i * TPB + t;
            int r = e >> 5, kk = e & 31;
            Atile[r][kk] = (_Float16)Ag[(size_t)r * DF + k0 + kk];
        }
        // stage pre-packed W chunk: pure 16B copies, no conversion
        {
            const H8* src  = reinterpret_cast<const H8*>(Wh + (size_t)(k0 >> 5) * (16 * 32 * 16));
            H8*       dstB = reinterpret_cast<H8*>(&Bt[0][0][0]);
            #pragma unroll
            for (int i = 0; i < 4; ++i) dstB[t * 4 + i] = src[t * 4 + i];
        }
        __syncthreads();

        // prefetch next A chunk (global_prefetch_b8) while WMMAs run
        if (k0 + 32 < DF && t < 64) {
            int r = t >> 1, half = t & 1;
            __builtin_prefetch(&Ag[(size_t)r * DF + k0 + 32 + half * 16], 0, 3);
        }

        // A fragment (contiguous k order: k = 16*(lane/16) + e, matching packed B)
        v16h a;
        {
            const H8* alo = reinterpret_cast<const H8*>(&Atile[16 * rblk + m][2 * khalf]);
            const H8* ahi = reinterpret_cast<const H8*>(&Atile[16 * rblk + m][2 * khalf + 8]);
            #pragma unroll
            for (int i = 0; i < 8; ++i) { a[i] = alo->h[i]; a[8 + i] = ahi->h[i]; }
        }
        #pragma unroll
        for (int ci = 0; ci < 4; ++ci) {
            const H8* b0 = reinterpret_cast<const H8*>(&Bt[ctbase + ci][lane][0]);
            const H8* b1 = reinterpret_cast<const H8*>(&Bt[ctbase + ci][lane][8]);
            v16h bb;
            #pragma unroll
            for (int i = 0; i < 8; ++i) { bb[i] = b0->h[i]; bb[8 + i] = b1->h[i]; }
            acc[ci] = __builtin_amdgcn_wmma_f32_16x16x32_f16(
                false, a, false, bb, (short)0, acc[ci], false, false);
        }
    }
    __syncthreads();

    // C/D layout: VGPR j, lane L -> row = j + 8*(L>=16), col = L%16
    #pragma unroll
    for (int ci = 0; ci < 4; ++ci) {
        #pragma unroll
        for (int j = 0; j < 8; ++j) {
            int row = 16 * rblk + j + ((lane >> 4) << 3);
            int col = 16 * (ctbase + ci) + (lane & 15);
            featLDS[row][col] = acc[ci][j] + biasLDS[col];
        }
    }
    __syncthreads();

    // ---- per-row LN + GELU + L2 + sim + softmax (wave handles 4 rows) ----
    float* featOut   = out + FEAT_OFF;
    float* assignOut = out + AS_OFF;

    #pragma unroll
    for (int q = 0; q < 4; ++q) {
        int row = wv * 4 + q;
        int n   = n0 + row;
        int c0  = lane * 8;
        float x[8];
        #pragma unroll
        for (int j = 0; j < 8; ++j) x[j] = featLDS[row][c0 + j];
        float s = 0.f, s2 = 0.f;
        #pragma unroll
        for (int j = 0; j < 8; ++j) { s += x[j]; s2 += x[j] * x[j]; }
        s  = wave_allreduce(s);
        s2 = wave_allreduce(s2);
        float mean = s * (1.0f / DP);
        float var  = s2 * (1.0f / DP) - mean * mean;
        float rstd = rsqrtf(var + 1e-5f);
        float g[8]; float sg = 0.f;
        #pragma unroll
        for (int j = 0; j < 8; ++j) {
            float ln = (x[j] - mean) * rstd * g1LDS[c0 + j] + b1LDS[c0 + j];
            float gv = gelu_exact(ln);
            g[j] = gv; sg += gv * gv;
        }
        sg = wave_allreduce(sg);
        float nrm = fmaxf(sqrtf(sg), 1e-12f);

        #pragma unroll
        for (int j = 0; j < 8; ++j) featLDS[row][c0 + j] = g[j];
        float* fo = featOut + ((size_t)(b * N_ + n)) * DP + c0;
        #pragma unroll
        for (int j = 0; j < 8; ++j) fo[j] = g[j];

        float sim[KP];
        #pragma unroll
        for (int k = 0; k < KP; ++k) {
            float p = 0.f;
            #pragma unroll
            for (int j = 0; j < 8; ++j) p += g[j] * protoLDS[k][c0 + j];
            p = wave_allreduce(p);
            sim[k] = p / (nrm * 0.07f);
        }
        float mx = sim[0];
        #pragma unroll
        for (int k = 1; k < KP; ++k) mx = fmaxf(mx, sim[k]);
        float es = 0.f; float aW[KP];
        #pragma unroll
        for (int k = 0; k < KP; ++k) { float e = __expf(sim[k] - mx); aW[k] = e; es += e; }
        float inv = 1.0f / es;
        if (lane == 0) {
            float* ao = assignOut + ((size_t)(b * N_ + n)) * KP;
            #pragma unroll
            for (int k = 0; k < KP; ++k) {
                float av = aW[k] * inv;
                ao[k] = av;
                assignLDS[row][k] = av;
            }
        }
    }
    __syncthreads();

    // ---- block-local assign^T @ feat, mass, pos; atomic into ws ----
    {
        float accn[KP];
        #pragma unroll
        for (int k = 0; k < KP; ++k) accn[k] = 0.f;
        for (int r = 0; r < ROWS; ++r) {
            float fv = featLDS[r][t];
            #pragma unroll
            for (int k = 0; k < KP; ++k) accn[k] += assignLDS[r][k] * fv;
        }
        #pragma unroll
        for (int k = 0; k < KP; ++k)
            atomicAdd(&ws[WS_FNUM + ((size_t)(b * KP + k)) * DP + t], accn[k]);
    }
    if (t < KP) {
        float sm = 0.f;
        for (int r = 0; r < ROWS; ++r) sm += assignLDS[r][t];
        atomicAdd(&ws[WS_MASS + b * KP + t], sm);
    } else if (t < KP + 2 * KP) {
        int idx = t - KP; int k = idx >> 1; int c = idx & 1;
        float sm = 0.f;
        for (int r = 0; r < ROWS; ++r) {
            int n = n0 + r;
            float coord = (c == 0) ? (float)(n % Ww) * inv_w : (float)(n / Ww) * inv_h;
            sm += assignLDS[r][k] * coord;
        }
        atomicAdd(&ws[WS_POS + (b * KP + k) * 2 + c], sm);
    }
}

// -----------------------------------------------------------------------------
// Finalize: divide by mass, part_pos, refine MLP (+residual), salience head.
// One workgroup per (b,k) row. Tiny: 512 x ~0.4 MFLOP.
// -----------------------------------------------------------------------------
__global__ __launch_bounds__(256)
void finalize_kernel(const float* __restrict__ g2, const float* __restrict__ beta2,
                     const float* __restrict__ Wr1, const float* __restrict__ br1,
                     const float* __restrict__ Wr2, const float* __restrict__ br2,
                     const float* __restrict__ Ws1, const float* __restrict__ bs1,
                     const float* __restrict__ Ws2, const float* __restrict__ bs2,
                     const float* __restrict__ ws, float* __restrict__ out)
{
    __shared__ float pf[DP], hh[DP], t1[2 * DP], pfo[DP], red1[256], red2[256], sl[64];
    const int t  = threadIdx.x;
    const int bk = blockIdx.x;   // b*8 + k

    float mass = fmaxf(ws[WS_MASS + bk], 1e-6f);
    float pv   = ws[WS_FNUM + (size_t)bk * DP + t] / mass;
    pf[t] = pv;
    if (t < 2) out[POS_OFF + bk * 2 + t] = ws[WS_POS + bk * 2 + t] / mass;
    red1[t] = pv; red2[t] = pv * pv;
    __syncthreads();
    for (int sft = 128; sft > 0; sft >>= 1) {
        if (t < sft) { red1[t] += red1[t + sft]; red2[t] += red2[t + sft]; }
        __syncthreads();
    }
    float mean = red1[0] * (1.0f / DP);
    float var  = red2[0] * (1.0f / DP) - mean * mean;
    float rstd = rsqrtf(var + 1e-5f);
    hh[t] = (pv - mean) * rstd * g2[t] + beta2[t];
    __syncthreads();

    #pragma unroll
    for (int half = 0; half < 2; ++half) {
        int j = t + half * 256;
        float acc = br1[j];
        for (int d = 0; d < DP; ++d) acc += hh[d] * Wr1[(size_t)d * (2 * DP) + j];
        t1[j] = gelu_exact(acc);
    }
    __syncthreads();
    {
        float acc = br2[t];
        for (int j = 0; j < 2 * DP; ++j) acc += t1[j] * Wr2[(size_t)j * DP + t];
        float v = pf[t] + acc;
        pfo[t] = v;
        out[PF_OFF + (size_t)bk * DP + t] = v;
    }
    __syncthreads();
    if (t < 64) {
        float acc = bs1[t];
        for (int d = 0; d < DP; ++d) acc += pfo[d] * Ws1[(size_t)d * 64 + t];
        sl[t] = gelu_exact(acc);
    }
    __syncthreads();
    if (t == 0) {
        float acc = bs2[0];
        for (int j = 0; j < 64; ++j) acc += sl[j] * Ws2[j];
        out[SAL_OFF + bk] = 1.0f / (1.0f + __expf(-acc));
    }
}

extern "C" void kernel_launch(void* const* d_in, const int* in_sizes, int n_in,
                              void* d_out, int out_size, void* d_ws, size_t ws_size,
                              hipStream_t stream)
{
    const float* patch = (const float*)d_in[0];
    const int*   Hp    = (const int*)d_in[1];
    const int*   Wpd   = (const int*)d_in[2];
    const float* Wproj = (const float*)d_in[3];
    const float* bproj = (const float*)d_in[4];
    const float* g1    = (const float*)d_in[5];
    const float* beta1 = (const float*)d_in[6];
    const float* proto = (const float*)d_in[7];
    const float* g2    = (const float*)d_in[8];
    const float* beta2 = (const float*)d_in[9];
    const float* Wr1   = (const float*)d_in[10];
    const float* br1   = (const float*)d_in[11];
    const float* Wr2   = (const float*)d_in[12];
    const float* br2   = (const float*)d_in[13];
    const float* Ws1   = (const float*)d_in[14];
    const float* bs1   = (const float*)d_in[15];
    const float* Ws2   = (const float*)d_in[16];
    const float* bs2   = (const float*)d_in[17];
    float* out = (float*)d_out;
    float* ws  = (float*)d_ws;
    const _Float16* Wh = (const _Float16*)(ws + WS_WH);

    zero_ws_kernel<<<(WS_ACC_TOTAL + 255) / 256, 256, 0, stream>>>(ws);
    pack_w_kernel<<<(DF * DP + 255) / 256, 256, 0, stream>>>(Wproj, ws);
    fused_feat_kernel<<<(B_ * N_) / ROWS, TPB, 0, stream>>>(
        patch, Hp, Wpd, Wh, bproj, g1, beta1, proto, out, ws);
    finalize_kernel<<<B_ * KP, 256, 0, stream>>>(
        g2, beta2, Wr1, br1, Wr2, br2, Ws1, bs1, Ws2, bs2, ws, out);
}
